// REGBlock_34230889349691
// MI455X (gfx1250) — compile-verified, matrix operand order checked by reading, not run on hardware
//
#include <hip/hip_runtime.h>

// ---------------------------------------------------------------------------
// CDNA5 WMMA types (wave32): v_wmma_f32_16x16x32_bf16
//   A: 16x32 bf16 (v16bf/lane), B: 32x16 bf16 (v16bf/lane), C/D: 16x16 f32 (v8f)
// ---------------------------------------------------------------------------
typedef __attribute__((ext_vector_type(16))) __bf16 v16bf;
typedef __attribute__((ext_vector_type(8)))  float  v8f;

// error-compensated fp32 -> bf16 hi/lo split (~16 mantissa bits total)
__device__ inline void split1(float f, __bf16* h, __bf16* l) {
  __bf16 hh = (__bf16)f;
  *h = hh;
  *l = (__bf16)(f - (float)hh);
}

union BFrag { v16bf v; uint4 q[2]; };

// Fragment load per documented 16-bit A/B layout:
//  lane<16 : row = lane,    elements = K {kb..kb+7, kb+16..kb+23} with kb=0
//  lane>=16: row = lane-16, same with kb=8
__device__ inline v16bf ld_frag(const __bf16* s, int row, int kb) {
  BFrag f;
  f.q[0] = *(const uint4*)(s + row * 40 + kb);
  f.q[1] = *(const uint4*)(s + row * 40 + kb + 16);
  return f.v;
}

#define WMMA_BF16(A, B, C) \
  __builtin_amdgcn_wmma_f32_16x16x32_bf16(false, (A), false, (B), (short)0, (C), false, false)

// ---------------------------------------------------------------------------
// bf16x3 GEMM:  C = epilogue( (Ah+Al)[M,K] @ (Bh+Bl)^T )   (B given as [Nc,K])
//   hi*hi + hi*lo + lo*hi on v_wmma_f32_16x16x32_bf16, fp32 accumulate.
// 128 threads / 4 waves; workgroup tile 64x64; each wave 32x32 (2x2 accs,
// fragments reused twice -> 12 WMMA per 16 ds_load_b128 per k-step);
// TK=32, double-buffered LDS, register-prefetch software pipeline.
// M,K,Nc multiples of 64/32 in every call here.
// Outputs (nullable): Cf fp32; Ch/Cl bf16 pair, optionally transposed with
// batch fold 2^sbShift (used to produce v^T for the attention GEMM).
// gram=1: C = split( 1 / (1 + max(sq[row]+sq[col]-2*acc,0)/temp^2) )
// ---------------------------------------------------------------------------
__global__ __launch_bounds__(128) void gemm_bf16x3(
    const __bf16* __restrict__ Ah, const __bf16* __restrict__ Al,
    const __bf16* __restrict__ Bh, const __bf16* __restrict__ Bl,
    const float* __restrict__ bias, const float* __restrict__ resid,
    float* __restrict__ Cf, __bf16* __restrict__ Ch, __bf16* __restrict__ Cl,
    int M, int K, int Nc,
    long sA, long sB, long sC, long sR,
    int outT, int sbShift,
    int gram, const float* __restrict__ sq, long sSq,
    const float* __restrict__ ltp)
{
  __shared__ __align__(16) __bf16 sAh[2][64 * 40];
  __shared__ __align__(16) __bf16 sAl[2][64 * 40];
  __shared__ __align__(16) __bf16 sBh[2][64 * 40];
  __shared__ __align__(16) __bf16 sBl[2][64 * 40];

  const int z = blockIdx.z;
  Ah += (size_t)z * sA;  Al += (size_t)z * sA;
  Bh += (size_t)z * sB;  Bl += (size_t)z * sB;
  if (Cf) Cf += (size_t)z * sC;
  if (Ch) { Ch += (size_t)z * sC; Cl += (size_t)z * sC; }
  if (resid) resid += (size_t)z * sR;

  const float* sqz = gram ? (sq + (size_t)z * sSq) : nullptr;
  float invT2 = 0.f;
  if (gram) {
    float t = fmaxf(expf(ltp[0]), 1e-5f);
    invT2 = 1.f / (t * t);
  }

  const int n0   = blockIdx.x * 64;
  const int m0   = blockIdx.y * 64;
  const int tid  = threadIdx.x;
  const int lane = tid & 31;
  const int wave = tid >> 5;
  const int wm   = (wave >> 1) * 32;   // wave's 32-row block in the 64x64 tile
  const int wn   = (wave & 1) * 32;    // wave's 32-col block

  // staging: each thread moves two uint4 (rows r and r+32) per operand array
  const int sr = tid >> 2;        // 0..31
  const int sc = (tid & 3) * 8;   // 0,8,16,24
  const __bf16* gAh = Ah + (size_t)(m0 + sr) * K + sc;
  const __bf16* gAl = Al + (size_t)(m0 + sr) * K + sc;
  const __bf16* gBh = Bh + (size_t)(n0 + sr) * K + sc;
  const __bf16* gBl = Bl + (size_t)(n0 + sr) * K + sc;
  const size_t gstep = (size_t)32 * K;      // +32 rows
  const int sidx  = sr * 40 + sc;
  const int sidx2 = sidx + 32 * 40;

  uint4 pa0, pa1, pal0, pal1, pb0, pb1, pbl0, pbl1;
  // prologue: stage k-tile 0 into buffer 0
  pa0  = *(const uint4*)gAh;            pa1  = *(const uint4*)(gAh + gstep);
  pal0 = *(const uint4*)gAl;            pal1 = *(const uint4*)(gAl + gstep);
  pb0  = *(const uint4*)gBh;            pb1  = *(const uint4*)(gBh + gstep);
  pbl0 = *(const uint4*)gBl;            pbl1 = *(const uint4*)(gBl + gstep);
  *(uint4*)(&sAh[0][sidx]) = pa0;       *(uint4*)(&sAh[0][sidx2]) = pa1;
  *(uint4*)(&sAl[0][sidx]) = pal0;      *(uint4*)(&sAl[0][sidx2]) = pal1;
  *(uint4*)(&sBh[0][sidx]) = pb0;       *(uint4*)(&sBh[0][sidx2]) = pb1;
  *(uint4*)(&sBl[0][sidx]) = pbl0;      *(uint4*)(&sBl[0][sidx2]) = pbl1;
  __syncthreads();

  v8f a00 = {0.f,0.f,0.f,0.f,0.f,0.f,0.f,0.f};
  v8f a01 = a00, a10 = a00, a11 = a00;

  const int kb  = (lane & 16) ? 8 : 0;
  const int rA0 = wm + (lane & 15), rA1 = rA0 + 16;
  const int rB0 = wn + (lane & 15), rB1 = rB0 + 16;

  const int nk = K >> 5;
  for (int kt = 0; kt < nk; ++kt) {
    const int cur = kt & 1, nxt = cur ^ 1;
    const int kn = (kt + 1 < nk) ? ((kt + 1) << 5) : 0;  // wrap: harmless reload

    // prefetch next k-tile into registers (overlaps with compute below)
    pa0  = *(const uint4*)(gAh + kn);   pa1  = *(const uint4*)(gAh + gstep + kn);
    pal0 = *(const uint4*)(gAl + kn);   pal1 = *(const uint4*)(gAl + gstep + kn);
    pb0  = *(const uint4*)(gBh + kn);   pb1  = *(const uint4*)(gBh + gstep + kn);
    pbl0 = *(const uint4*)(gBl + kn);   pbl1 = *(const uint4*)(gBl + gstep + kn);

    // compute on current buffer: 8 fragments, 12 WMMAs (each fragment used 2x)
    v16bf fah0 = ld_frag(sAh[cur], rA0, kb);
    v16bf fah1 = ld_frag(sAh[cur], rA1, kb);
    v16bf fal0 = ld_frag(sAl[cur], rA0, kb);
    v16bf fal1 = ld_frag(sAl[cur], rA1, kb);
    v16bf fbh0 = ld_frag(sBh[cur], rB0, kb);
    v16bf fbh1 = ld_frag(sBh[cur], rB1, kb);
    v16bf fbl0 = ld_frag(sBl[cur], rB0, kb);
    v16bf fbl1 = ld_frag(sBl[cur], rB1, kb);

    a00 = WMMA_BF16(fah0, fbh0, a00);
    a01 = WMMA_BF16(fah0, fbh1, a01);
    a10 = WMMA_BF16(fah1, fbh0, a10);
    a11 = WMMA_BF16(fah1, fbh1, a11);
    a00 = WMMA_BF16(fah0, fbl0, a00);
    a01 = WMMA_BF16(fah0, fbl1, a01);
    a10 = WMMA_BF16(fah1, fbl0, a10);
    a11 = WMMA_BF16(fah1, fbl1, a11);
    a00 = WMMA_BF16(fal0, fbh0, a00);
    a01 = WMMA_BF16(fal0, fbh1, a01);
    a10 = WMMA_BF16(fal1, fbh0, a10);
    a11 = WMMA_BF16(fal1, fbh1, a11);

    // drain prefetch registers into the other buffer
    *(uint4*)(&sAh[nxt][sidx]) = pa0;   *(uint4*)(&sAh[nxt][sidx2]) = pa1;
    *(uint4*)(&sAl[nxt][sidx]) = pal0;  *(uint4*)(&sAl[nxt][sidx2]) = pal1;
    *(uint4*)(&sBh[nxt][sidx]) = pb0;   *(uint4*)(&sBh[nxt][sidx2]) = pb1;
    *(uint4*)(&sBl[nxt][sidx]) = pbl0;  *(uint4*)(&sBl[nxt][sidx2]) = pbl1;
    __syncthreads();
  }

  // epilogue: acc element r -> row (+ i*16 + 8*(lane>=16) + r), col (+ j*16 + lane&15)
  v8f accs[2][2] = {{a00, a01}, {a10, a11}};
#pragma unroll
  for (int i = 0; i < 2; ++i) {
    const int rbase = m0 + wm + i * 16 + ((lane & 16) ? 8 : 0);
#pragma unroll
    for (int j = 0; j < 2; ++j) {
      const int col = n0 + wn + j * 16 + (lane & 15);
      const float bcol = (!gram && bias) ? bias[col] : 0.f;
      const float sqc  = gram ? sqz[col] : 0.f;
#pragma unroll
      for (int r = 0; r < 8; ++r) {
        const int row = rbase + r;
        const size_t ci = (size_t)row * Nc + col;
        float val = accs[i][j][r];
        if (gram) {
          float d2 = fmaxf(sqz[row] + sqc - 2.f * val, 0.f);  // dist^2, clamped
          val = 1.f / (1.f + d2 * invT2);                     // resonance kernel
          split1(val, &Ch[ci], &Cl[ci]);
        } else {
          val += bcol;
          if (resid) val += resid[ci];
          if (Cf) Cf[ci] = val;
          if (Ch) {
            size_t oi = outT
                ? ((size_t)(row >> sbShift) * ((size_t)Nc << sbShift)
                   + ((size_t)col << sbShift) + (size_t)(row & ((1 << sbShift) - 1)))
                : ci;
            split1(val, &Ch[oi], &Cl[oi]);
          }
        }
      }
    }
  }
}

// ---------------------------------------------------------------------------
// Elementwise / reduction helpers
// ---------------------------------------------------------------------------
__device__ inline float block_sum(float v, float* red) {
  int t = threadIdx.x;
  red[t] = v;
  __syncthreads();
  for (int s = 128; s > 0; s >>= 1) {
    if (t < s) red[t] += red[t + s];
    __syncthreads();
  }
  float r = red[0];
  __syncthreads();
  return r;
}

// fp32 -> bf16 hi/lo split (flat)
__global__ __launch_bounds__(256) void split_kernel(const float* __restrict__ in,
                                                    __bf16* __restrict__ h,
                                                    __bf16* __restrict__ l, long n) {
  long i = (long)blockIdx.x * 256 + threadIdx.x;
  if (i < n) split1(in[i], &h[i], &l[i]);
}

// fp32 [R,C] -> transposed bf16 hi/lo [C,R]; blockIdx.y = batch (stride R*C)
__global__ __launch_bounds__(256) void tsplit_kernel(const float* __restrict__ in,
                                                     __bf16* __restrict__ h,
                                                     __bf16* __restrict__ l,
                                                     int R, int C) {
  size_t base = (size_t)blockIdx.y * R * C;
  long i = (long)blockIdx.x * 256 + threadIdx.x;
  if (i < (long)R * C) {
    int c = (int)(i / R), r = (int)(i % R);
    split1(in[base + (size_t)r * C + c], &h[base + i], &l[base + i]);
  }
}

// sq[row] = |x[row,:]|^2
__global__ __launch_bounds__(256) void sq_kernel(const float* __restrict__ x,
                                                 float* __restrict__ sq, int D) {
  __shared__ float red[256];
  size_t base = (size_t)blockIdx.x * D;
  float s = 0.f;
  for (int d = threadIdx.x; d < D; d += 256) { float v = x[base + d]; s += v * v; }
  s = block_sum(s, red);
  if (threadIdx.x == 0) sq[blockIdx.x] = s;
}

// attn = k / max(sum_t k, 1e-8), operating on the bf16 hi/lo pair in place
__global__ __launch_bounds__(256) void rownorm_hl(__bf16* __restrict__ hi,
                                                  __bf16* __restrict__ lo, int S) {
  __shared__ float red[256];
  size_t base = (size_t)blockIdx.x * S;
  float s = 0.f;
  for (int t = threadIdx.x; t < S; t += 256)
    s += (float)hi[base + t] + (float)lo[base + t];
  s = block_sum(s, red);
  float r = 1.f / fmaxf(s, 1e-8f);
  for (int t = threadIdx.x; t < S; t += 256) {
    float f = ((float)hi[base + t] + (float)lo[base + t]) * r;
    split1(f, &hi[base + t], &lo[base + t]);
  }
}

// out = LN(in)*g+b (eps 1e-5); optional fused bf16 hi/lo copy of the output
__global__ __launch_bounds__(256) void ln_kernel(const float* __restrict__ in,
                                                 const float* __restrict__ g,
                                                 const float* __restrict__ b,
                                                 float* __restrict__ out,
                                                 __bf16* __restrict__ oh,
                                                 __bf16* __restrict__ ol, int D) {
  __shared__ float red[256];
  size_t base = (size_t)blockIdx.x * D;
  float s = 0.f, s2 = 0.f;
  for (int d = threadIdx.x; d < D; d += 256) { float v = in[base + d]; s += v; s2 += v * v; }
  s  = block_sum(s, red);
  s2 = block_sum(s2, red);
  float mean = s / D;
  float rstd = rsqrtf(s2 / D - mean * mean + 1e-5f);
  for (int d = threadIdx.x; d < D; d += 256) {
    float v = (in[base + d] - mean) * rstd * g[d] + b[d];
    if (out) out[base + d] = v;
    if (oh) split1(v, &oh[base + d], &ol[base + d]);
  }
}

// soma (+)= w[n] * tanh(LN(pre)*lng+lnb) * sigmoid(gpre)
__global__ __launch_bounds__(256) void dend_kernel(const float* __restrict__ pre,
                                                   const float* __restrict__ gpre,
                                                   const float* __restrict__ lng,
                                                   const float* __restrict__ lnb,
                                                   const float* __restrict__ plog,
                                                   int n, float* __restrict__ soma, int H) {
  __shared__ float red[256];
  size_t base = (size_t)blockIdx.x * H;
  float l0 = plog[0], l1 = plog[1], l2 = plog[2], l3 = plog[3];
  float mx = fmaxf(fmaxf(l0, l1), fmaxf(l2, l3));
  float e0 = expf(l0 - mx), e1 = expf(l1 - mx), e2 = expf(l2 - mx), e3 = expf(l3 - mx);
  float wn = ((n == 0) ? e0 : (n == 1) ? e1 : (n == 2) ? e2 : e3) / (e0 + e1 + e2 + e3);

  float s = 0.f, s2 = 0.f;
  for (int h = threadIdx.x; h < H; h += 256) { float v = pre[base + h]; s += v; s2 += v * v; }
  s  = block_sum(s, red);
  s2 = block_sum(s2, red);
  float mean = s / H;
  float rstd = rsqrtf(s2 / H - mean * mean + 1e-5f);
  for (int h = threadIdx.x; h < H; h += 256) {
    float dv = tanhf((pre[base + h] - mean) * rstd * lng[h] + lnb[h]);
    float gv = 1.f / (1.f + expf(-gpre[base + h]));
    float val = wn * dv * gv;
    soma[base + h] = (n == 0) ? val : (soma[base + h] + val);
  }
}

// homeostatic norm: membrane -> d_out; act = tanh(membrane) as bf16 hi/lo
__global__ __launch_bounds__(256) void finalize_kernel(const float* __restrict__ soma,
                                                       const float* __restrict__ scale,
                                                       float* __restrict__ memb_out,
                                                       __bf16* __restrict__ acth,
                                                       __bf16* __restrict__ actl, int H) {
  __shared__ float red[256];
  size_t base = (size_t)blockIdx.x * H;
  float s = 0.f, s2 = 0.f;
  for (int h = threadIdx.x; h < H; h += 256) {
    float v = soma[base + h] * scale[h];
    s += v; s2 += v * v;
  }
  s  = block_sum(s, red);
  s2 = block_sum(s2, red);
  float mean = s / H;
  float rstd = rsqrtf(s2 / H - mean * mean + 1e-5f);
  for (int h = threadIdx.x; h < H; h += 256) {
    float m = (soma[base + h] * scale[h] - mean) * rstd;
    memb_out[base + h] = m;
    split1(tanhf(m), &acth[base + h], &actl[base + h]);
  }
}

// ---------------------------------------------------------------------------
// Orchestration.  B=4, S=2048, D=512, H=1024, N=4.
// ---------------------------------------------------------------------------
extern "C" void kernel_launch(void* const* d_in, const int* in_sizes, int n_in,
                              void* d_out, int out_size, void* d_ws, size_t ws_size,
                              hipStream_t stream) {
  (void)in_sizes; (void)n_in; (void)out_size; (void)ws_size;
  const int B = 4, S = 2048, D = 512, H = 1024;
  const int M = B * S;  // 8192

  const float* x   = (const float*)d_in[0];
  const float* Wv  = (const float*)d_in[1];
  const float* bv  = (const float*)d_in[2];
  const float* ltp = (const float*)d_in[3];
  const float* g1  = (const float*)d_in[4];
  const float* b1  = (const float*)d_in[5];
  const float* Wd  = (const float*)d_in[6];
  const float* bd  = (const float*)d_in[7];
  const float* lng = (const float*)d_in[8];
  const float* lnb = (const float*)d_in[9];
  const float* Wg  = (const float*)d_in[10];
  const float* bg  = (const float*)d_in[11];
  const float* pl  = (const float*)d_in[12];
  const float* ss  = (const float*)d_in[13];
  const float* Wo  = (const float*)d_in[14];
  const float* bo  = (const float*)d_in[15];
  const float* g2  = (const float*)d_in[16];
  const float* b2  = (const float*)d_in[17];

  float* x2out = (float*)d_out;
  float* membO = x2out + (size_t)M * D;

  // ---- workspace carve (256B aligned), ~200 MB ----
  char* wp = (char*)d_ws;
  auto alloc = [&](size_t bytes) { char* p = wp; wp += (bytes + 255) & ~(size_t)255; return p; };
  float*  sqb  = (float*)alloc((size_t)M * 4);
  float*  tmpb = (float*)alloc((size_t)M * D * 4);
  float*  x1f  = (float*)alloc((size_t)M * D * 4);
  float*  somab= (float*)alloc((size_t)M * H * 4);
  __bf16* x1h  = (__bf16*)alloc((size_t)M * D * 2);
  __bf16* x1l  = (__bf16*)alloc((size_t)M * D * 2);
  __bf16* WvTh = (__bf16*)alloc((size_t)D * D * 2);
  __bf16* WvTl = (__bf16*)alloc((size_t)D * D * 2);
  __bf16* WdTh = (__bf16*)alloc((size_t)4 * H * D * 2);
  __bf16* WdTl = (__bf16*)alloc((size_t)4 * H * D * 2);
  __bf16* WgTh = (__bf16*)alloc((size_t)4 * H * D * 2);
  __bf16* WgTl = (__bf16*)alloc((size_t)4 * H * D * 2);
  __bf16* WoTh = (__bf16*)alloc((size_t)H * D * 2);
  __bf16* WoTl = (__bf16*)alloc((size_t)H * D * 2);
  // region R2 (32MB): x hi/lo + v^T hi/lo; reused later for act hi/lo
  char* R2 = alloc((size_t)M * D * 2 * 4);
  __bf16* xh   = (__bf16*)R2;
  __bf16* xl   = xh + (size_t)M * D;
  __bf16* vTh  = xl + (size_t)M * D;
  __bf16* vTl  = vTh + (size_t)M * D;
  __bf16* acth = (__bf16*)R2;                 // M*H each; exactly fills R2
  __bf16* actl = acth + (size_t)M * H;
  // region R1 (67MB): attention kernel hi/lo; reused later for pre/gpre f32
  size_t KbE = (size_t)B * S * S;
  char* R1 = alloc(KbE * 4);
  __bf16* Kbh  = (__bf16*)R1;
  __bf16* Kbl  = Kbh + KbE;
  float*  preb = (float*)R1;
  float*  gpreb= preb + (size_t)M * H;

  const long z0 = 0;

  // 0) one-time splits: x -> hi/lo; weights -> transposed hi/lo
  {
    long n = (long)M * D;
    split_kernel<<<(unsigned)((n + 255) / 256), 256, 0, stream>>>(x, xh, xl, n);
    tsplit_kernel<<<dim3((D * D + 255) / 256, 1), 256, 0, stream>>>(Wv, WvTh, WvTl, D, D);
    tsplit_kernel<<<dim3((D * H + 255) / 256, 4), 256, 0, stream>>>(Wd, WdTh, WdTl, D, H);
    tsplit_kernel<<<dim3((D * H + 255) / 256, 4), 256, 0, stream>>>(Wg, WgTh, WgTl, D, H);
    tsplit_kernel<<<dim3((H * D + 255) / 256, 1), 256, 0, stream>>>(Wo, WoTh, WoTl, H, D);
  }

  // 1) sq[b,s] = |x|^2
  sq_kernel<<<M, 256, 0, stream>>>(x, sqb, D);

  // 2) v = x@Wv + bv, written directly as v^T hi/lo per batch ([B][D][S])
  gemm_bf16x3<<<dim3(D / 64, M / 64, 1), 128, 0, stream>>>(
      xh, xl, WvTh, WvTl, bv, nullptr, nullptr, vTh, vTl,
      M, D, D, z0, z0, z0, z0, /*outT=*/1, /*sbShift=*/11, 0, nullptr, 0, nullptr);

  // 3) resonance kernel matrix (Gram + rational kernel), bf16 hi/lo, per batch
  gemm_bf16x3<<<dim3(S / 64, S / 64, B), 128, 0, stream>>>(
      xh, xl, xh, xl, nullptr, nullptr, nullptr, Kbh, Kbl,
      S, D, S, (long)S * D, (long)S * D, (long)S * S, z0, 0, 0,
      /*gram=*/1, sqb, (long)S, ltp);

  // 4) row-normalize attention weights in place (hi/lo)
  rownorm_hl<<<M, 256, 0, stream>>>(Kbh, Kbl, S);

  // 5) tmp = x + attn @ v            (A=[S,S] hl, B=v^T [D,S] hl, per batch)
  gemm_bf16x3<<<dim3(D / 64, S / 64, B), 128, 0, stream>>>(
      Kbh, Kbl, vTh, vTl, nullptr, x, tmpb, nullptr, nullptr,
      S, S, D, (long)S * S, (long)D * S, (long)S * D, (long)S * D, 0, 0,
      0, nullptr, 0, nullptr);

  // 6) x1 = LN(tmp; g1,b1), fused hi/lo split
  ln_kernel<<<M, 256, 0, stream>>>(tmpb, g1, b1, x1f, x1h, x1l, D);

  // 7) dendrites, sequential over n (pre/gpre reuse the kernel-matrix region)
  for (int n = 0; n < 4; ++n) {
    gemm_bf16x3<<<dim3(H / 64, M / 64, 1), 128, 0, stream>>>(
        x1h, x1l, WdTh + (size_t)n * H * D, WdTl + (size_t)n * H * D,
        bd + (size_t)n * H, nullptr, preb, nullptr, nullptr,
        M, D, H, z0, z0, z0, z0, 0, 0, 0, nullptr, 0, nullptr);
    gemm_bf16x3<<<dim3(H / 64, M / 64, 1), 128, 0, stream>>>(
        x1h, x1l, WgTh + (size_t)n * H * D, WgTl + (size_t)n * H * D,
        bg + (size_t)n * H, nullptr, gpreb, nullptr, nullptr,
        M, D, H, z0, z0, z0, z0, 0, 0, 0, nullptr, 0, nullptr);
    dend_kernel<<<M, 256, 0, stream>>>(preb, gpreb, lng + (size_t)n * H,
                                       lnb + (size_t)n * H, pl, n, somab, H);
  }

  // 8) homeostatic norm -> membrane (output #2); act = tanh as bf16 hi/lo
  finalize_kernel<<<M, 256, 0, stream>>>(somab, ss, membO, acth, actl, H);

  // 9) tmp = x1 + act @ Wo + bo
  gemm_bf16x3<<<dim3(D / 64, M / 64, 1), 128, 0, stream>>>(
      acth, actl, WoTh, WoTl, bo, x1f, tmpb, nullptr, nullptr,
      M, H, D, z0, z0, z0, z0, 0, 0, 0, nullptr, 0, nullptr);

  // 10) x2 = LN(tmp; g2,b2) -> output #1
  ln_kernel<<<M, 256, 0, stream>>>(tmpb, g2, b2, x2out, nullptr, nullptr, D);
}
